// ElmanRNN_19292993094207
// MI455X (gfx1250) — matrix operand order repaired and checked by reading, not compiled, not run on hardware
//
#include <hip/hip_runtime.h>
#include <stdint.h>

// Problem constants (fixed by the reference).
#define DD 1024   // embedding dim
#define BB 32     // batch
#define SS 1024   // sequence length
#define LL 2      // layers

typedef __bf16 bf16x16 __attribute__((ext_vector_type(16)));
typedef float  f32x8   __attribute__((ext_vector_type(8)));

union FragU {
  uint4    q[2];
  unsigned u[8];
  bf16x16  v;
};

// RNE f32 -> bf16 raw bits.
__device__ __forceinline__ uint16_t f2bf_rne(float f) {
  unsigned u = __float_as_uint(f);
  unsigned r = u + 0x7FFFu + ((u >> 16) & 1u);
  return (uint16_t)(r >> 16);
}

// Hardware tanh if the gfx1250 builtin exists; libm fallback otherwise.
#if __has_builtin(__builtin_amdgcn_tanhf)
#define FAST_TANH(x) __builtin_amdgcn_tanhf(x)
#elif __has_builtin(__builtin_amdgcn_tanh_f32)
#define FAST_TANH(x) __builtin_amdgcn_tanh_f32(x)
#else
#define FAST_TANH(x) tanhf(x)
#endif

// B fragment (16x16x32 bf16): weights stored transposed as WT[n][k] bf16.
// Layout used here (contiguous k, same as A): lane half h, VGPR j -> K = 2j + {0,1} + 16h, N = lane%16.
// => 16 consecutive bf16 (32B) at column n, row k0+16h: two b128 loads.
__device__ __forceinline__ bf16x16 load_b_frag(const uint16_t* colbase, int k0, int half) {
  const uint16_t* p = colbase + k0 + 16 * half;
  FragU f;
  f.q[0] = *(const uint4*)(p);
  f.q[1] = *(const uint4*)(p + 8);
  return f.v;
}

// A fragment (16x16x32 bf16) from bf16 row-major activations.
// Layout: lane half h, row m = lane%16; VGPR j -> K = 16h + 2j + {0,1}
// (matches B fragment): 16 consecutive bf16 -> 2 b128.
__device__ __forceinline__ bf16x16 load_a_frag(const uint16_t* row, int k0, int half) {
  FragU f;
  f.q[0] = *(const uint4*)(row + k0 + 16 * half);
  f.q[1] = *(const uint4*)(row + k0 + 16 * half + 8);
  return f.v;
}

#define WMMA_BF16(A, B, C) \
  __builtin_amdgcn_wmma_f32_16x16x32_bf16(false, (A), false, (B), (short)0, (C), false, false)

// One (timestep, layer): h_new = tanh(inp @ Wh + h_prev @ Uh + b).
// M=32, N=1024, K=1024 twice (Wh and Uh fused in one K loop).
// Each wave owns a 32x32 output block (4 accumulators); per K-step:
// 4 B frags + 4 A frags (16 b128 loads) feed 8 chained WMMAs.
// K loop fully unrolled (32 steps): all load offsets become immediates,
// zero loop-carried address math / branches.
// Grid: 8 blocks x 128 threads (4 waves) = 32 waves = 32 N-superblocks.
__global__ void __launch_bounds__(128)
rnn_step_kernel(const uint16_t* __restrict__ inpA, long inp_stride,  // [32][*] bf16
                const uint16_t* __restrict__ hA,                     // [32][DD] bf16
                const uint16_t* __restrict__ WT,                     // [DD][DD] bf16^T
                const uint16_t* __restrict__ UT,                     // [DD][DD] bf16^T
                const float* __restrict__ bias,                      // [DD]
                uint16_t* __restrict__ hOut,                         // [32][DD] bf16
                float* __restrict__ seqOut, long seq_stride) {       // optional f32
  const int lane = threadIdx.x & 31;
  const int wave = threadIdx.x >> 5;
  const int half = lane >> 4;
  const int col  = lane & 15;
  const int n0   = (blockIdx.x * 4 + wave) * 32;   // 32-wide N superblock

  const uint16_t* inp_row0 = inpA + (size_t)(col) * inp_stride;        // M rows 0..15
  const uint16_t* inp_row1 = inpA + (size_t)(col + 16) * inp_stride;   // M rows 16..31
  const uint16_t* h_row0   = hA   + (size_t)(col) * DD;
  const uint16_t* h_row1   = hA   + (size_t)(col + 16) * DD;
  const uint16_t* wcol0    = WT   + (size_t)(n0 + col) * DD;
  const uint16_t* wcol1    = WT   + (size_t)(n0 + 16 + col) * DD;
  const uint16_t* ucol0    = UT   + (size_t)(n0 + col) * DD;
  const uint16_t* ucol1    = UT   + (size_t)(n0 + 16 + col) * DD;

  f32x8 acc00 = {0.f, 0.f, 0.f, 0.f, 0.f, 0.f, 0.f, 0.f};
  f32x8 acc01 = acc00, acc10 = acc00, acc11 = acc00;

#pragma unroll
  for (int k0 = 0; k0 < DD; k0 += 32) {
    bf16x16 bw0 = load_b_frag(wcol0, k0, half);
    bf16x16 bw1 = load_b_frag(wcol1, k0, half);
    bf16x16 bu0 = load_b_frag(ucol0, k0, half);
    bf16x16 bu1 = load_b_frag(ucol1, k0, half);
    bf16x16 ai0 = load_a_frag(inp_row0, k0, half);
    bf16x16 ai1 = load_a_frag(inp_row1, k0, half);
    bf16x16 ah0 = load_a_frag(h_row0, k0, half);
    bf16x16 ah1 = load_a_frag(h_row1, k0, half);
    acc00 = WMMA_BF16(ai0, bw0, acc00);
    acc00 = WMMA_BF16(ah0, bu0, acc00);
    acc01 = WMMA_BF16(ai0, bw1, acc01);
    acc01 = WMMA_BF16(ah0, bu1, acc01);
    acc10 = WMMA_BF16(ai1, bw0, acc10);
    acc10 = WMMA_BF16(ah1, bu0, acc10);
    acc11 = WMMA_BF16(ai1, bw1, acc11);
    acc11 = WMMA_BF16(ah1, bu1, acc11);
  }

  // C/D layout: VGPR r, lane -> M = r + 8*half, N = lane%16.
  const float bv0 = bias[n0 + col];
  const float bv1 = bias[n0 + 16 + col];
#pragma unroll
  for (int r = 0; r < 8; ++r) {
    const int mr = r + 8 * half;
    const float v00 = FAST_TANH(acc00[r] + bv0);   // M=mr,    N=n0+col
    const float v01 = FAST_TANH(acc01[r] + bv1);   // M=mr,    N=n0+16+col
    const float v10 = FAST_TANH(acc10[r] + bv0);   // M=mr+16, N=n0+col
    const float v11 = FAST_TANH(acc11[r] + bv1);   // M=mr+16, N=n0+16+col
    hOut[(size_t)mr * DD + n0 + col]             = f2bf_rne(v00);
    hOut[(size_t)mr * DD + n0 + 16 + col]        = f2bf_rne(v01);
    hOut[(size_t)(mr + 16) * DD + n0 + col]      = f2bf_rne(v10);
    hOut[(size_t)(mr + 16) * DD + n0 + 16 + col] = f2bf_rne(v11);
    if (seqOut) {
      seqOut[(size_t)mr * seq_stride + n0 + col]             = v00;
      seqOut[(size_t)mr * seq_stride + n0 + 16 + col]        = v01;
      seqOut[(size_t)(mr + 16) * seq_stride + n0 + col]      = v10;
      seqOut[(size_t)(mr + 16) * seq_stride + n0 + 16 + col] = v11;
    }
  }
}

// One-time per call: WT[l][n][k] = bf16(Wh[l][k][n]); same for Uh.
__global__ void cvt_transpose_kernel(const float* __restrict__ Wh,
                                     const float* __restrict__ Uh,
                                     uint16_t* __restrict__ WT,
                                     uint16_t* __restrict__ UT) {
  size_t idx = (size_t)blockIdx.x * blockDim.x + threadIdx.x;
  const size_t total = (size_t)LL * DD * DD;
  if (idx >= total) return;
  size_t l   = idx / ((size_t)DD * DD);
  size_t rem = idx % ((size_t)DD * DD);
  size_t n = rem / DD, k = rem % DD;
  size_t src = l * (size_t)DD * DD + k * DD + n;
  WT[idx] = f2bf_rne(Wh[src]);
  UT[idx] = f2bf_rne(Uh[src]);
}

// One-time per call: x (f32) -> bf16, 4 elements per thread, coalesced.
__global__ void cvt_x_kernel(const float* __restrict__ x, uint16_t* __restrict__ xb,
                             size_t n4) {
  size_t i = (size_t)blockIdx.x * blockDim.x + threadIdx.x;
  if (i >= n4) return;
  float4 f = *(const float4*)(x + i * 4);
  union { uint16_t h[4]; uint2 u; } o;
  o.h[0] = f2bf_rne(f.x);
  o.h[1] = f2bf_rne(f.y);
  o.h[2] = f2bf_rne(f.z);
  o.h[3] = f2bf_rne(f.w);
  *(uint2*)(xb + i * 4) = o.u;
}

__global__ void zero_kernel(uint16_t* __restrict__ p, size_t n) {
  size_t i = (size_t)blockIdx.x * blockDim.x + threadIdx.x;
  if (i < n) p[i] = 0;
}

extern "C" void kernel_launch(void* const* d_in, const int* in_sizes, int n_in,
                              void* d_out, int out_size, void* d_ws, size_t ws_size,
                              hipStream_t stream) {
  const float* x  = (const float*)d_in[0];  // [B,S,D]
  const float* Wh = (const float*)d_in[1];  // [L,D,D]
  const float* Uh = (const float*)d_in[2];  // [L,D,D]
  const float* b  = (const float*)d_in[3];  // [L,D]
  float* out = (float*)d_out;               // [B,S,D]

  // Workspace layout:
  //   WT  bf16 [L][D][D]    : 4 MB
  //   UT  bf16 [L][D][D]    : 4 MB
  //   xb  bf16 [B][S][D]    : 64 MB (x pre-converted once; HBM-read x only once)
  //   hbuf bf16 [L][2][B][D]: 256 KB (ping-pong hidden state)
  uint16_t* WT   = (uint16_t*)d_ws;
  uint16_t* UT   = WT + (size_t)LL * DD * DD;
  uint16_t* xb   = UT + (size_t)LL * DD * DD;
  uint16_t* hbuf = xb + (size_t)BB * SS * DD;

  {
    const size_t total = (size_t)LL * DD * DD;
    cvt_transpose_kernel<<<(unsigned)((total + 255) / 256), 256, 0, stream>>>(Wh, Uh, WT, UT);
  }
  {
    const size_t n4 = (size_t)BB * SS * DD / 4;
    cvt_x_kernel<<<(unsigned)((n4 + 255) / 256), 256, 0, stream>>>(x, xb, n4);
  }
  {
    const size_t n = (size_t)LL * 2 * BB * DD;
    zero_kernel<<<(unsigned)((n + 255) / 256), 256, 0, stream>>>(hbuf, n);
  }

  const long xstride = (long)SS * DD;
  for (int t = 0; t < SS; ++t) {
    const int rd = t & 1, wr = rd ^ 1;
    uint16_t* h0r = hbuf + (size_t)(0 * 2 + rd) * BB * DD;
    uint16_t* h0w = hbuf + (size_t)(0 * 2 + wr) * BB * DD;
    uint16_t* h1r = hbuf + (size_t)(1 * 2 + rd) * BB * DD;
    uint16_t* h1w = hbuf + (size_t)(1 * 2 + wr) * BB * DD;

    // Layer 0: inp = xb[:, t, :] (row stride S*D), no sequence output.
    rnn_step_kernel<<<8, 128, 0, stream>>>(
        xb + (size_t)t * DD, xstride, h0r,
        WT, UT, b, h0w, (float*)nullptr, 0L);

    // Layer 1: inp = layer-0 output (row stride D); writes out[:, t, :] f32.
    rnn_step_kernel<<<8, 128, 0, stream>>>(
        h0w, (long)DD, h1r,
        WT + (size_t)DD * DD, UT + (size_t)DD * DD, b + DD, h1w,
        out + (size_t)t * DD, (long)SS * DD);
  }
}
